// ScanNet72SemanticLoss_75084618268706
// MI455X (gfx1250) — matrix operand order repaired and checked by reading, not compiled, not run on hardware
//
#include <hip/hip_runtime.h>
#include <stdint.h>

typedef __attribute__((ext_vector_type(16))) _Float16 v16h;
typedef __attribute__((ext_vector_type(8)))  float    v8f;
typedef __attribute__((ext_vector_type(4)))  float    v4f;

#define NUM_C   72
#define PADC    80
#define DIM     32
#define NEGINF  (-1e30f)
#define INV_T   (1.0f / 0.07f)
#define TPW     4        // tiles per wave in the WMMA kernel

__device__ __forceinline__ bool is_missing(int c) {
    return (c == 13) | (c == 53) | (c == 61);
}

// ---------------- zero scratch ----------------
__global__ void k_zero(float* __restrict__ p, int n) {
    int i = blockIdx.x * blockDim.x + threadIdx.x;
    if (i < n) p[i] = 0.0f;
}

// ---------------- phase 1: segment sums of L2-normalized rows ----------------
__global__ __launch_bounds__(256) void k_segsum(
    const float* __restrict__ emb, const int* __restrict__ lab,
    float* __restrict__ gSums, float* __restrict__ gCnt, int N, int chunks) {
    __shared__ float sSums[NUM_C * DIM];
    __shared__ float sCnt[NUM_C];
    int b      = blockIdx.x / chunks;
    int chunk  = blockIdx.x % chunks;
    int rowsPer = N / chunks;
    int rowBase = chunk * rowsPer;
    int t = threadIdx.x;
    for (int i = t; i < NUM_C * DIM; i += blockDim.x) sSums[i] = 0.0f;
    for (int i = t; i < NUM_C; i += blockDim.x) sCnt[i] = 0.0f;
    __syncthreads();

    int lane = t & 31, wv = t >> 5, nw = blockDim.x >> 5;
    const float* eb = emb + (size_t)b * N * DIM;
    const int*   lb = lab + (size_t)b * N;

    for (int r = wv; r < rowsPer; r += nw) {
        int row = rowBase + r;
        int L = lb[row];
        float e = eb[(size_t)row * DIM + lane];
        float sq = e * e;
        #pragma unroll
        for (int m = 16; m >= 1; m >>= 1) sq += __shfl_xor(sq, m, 32);
        float scale = 1.0f / fmaxf(sqrtf(sq), 1e-12f);
        float val = e * scale;
        if (L >= 0) {
            int lc = min(L, NUM_C - 1);
            atomicAdd(&sSums[lc * DIM + lane], val);
            if (lane == 0) atomicAdd(&sCnt[lc], 1.0f);
        }
    }
    __syncthreads();
    for (int i = t; i < NUM_C * DIM; i += blockDim.x)
        atomicAdd(&gSums[(size_t)b * NUM_C * DIM + i], sSums[i]);
    for (int i = t; i < NUM_C; i += blockDim.x)
        atomicAdd(&gCnt[b * NUM_C + i], sCnt[i]);
}

// ---------------- phase 2: prototypes (f16, padded to 80 cols) + present bitmask ----
__global__ __launch_bounds__(256) void k_protos(
    const float* __restrict__ gSums, const float* __restrict__ gCnt,
    _Float16* __restrict__ protos, uint32_t* __restrict__ maskW) {
    int b = blockIdx.x;
    int t = threadIdx.x, lane = t & 31, wv = t >> 5, nw = blockDim.x >> 5;
    for (int c = wv; c < PADC; c += nw) {
        float v = 0.0f;
        bool pres = false;
        if (c < NUM_C) {
            float cnt = gCnt[b * NUM_C + c];
            v = gSums[(size_t)b * NUM_C * DIM + c * DIM + lane] / fmaxf(cnt, 1.0f);
            float sq = v * v;
            #pragma unroll
            for (int m = 16; m >= 1; m >>= 1) sq += __shfl_xor(sq, m, 32);
            v = v / fmaxf(sqrtf(sq), 1e-12f);
            pres = (cnt > 0.0f) && !is_missing(c);
        }
        protos[(size_t)b * PADC * DIM + c * DIM + lane] = (_Float16)v;
        if (lane == 0 && pres)
            atomicOr(&maskW[b * 3 + (c >> 5)], 1u << (c & 31));
    }
}

// ---------------- phase 3: WMMA logits + masked log-softmax NLL ----------------
__global__ __launch_bounds__(256) void k_loss(
    const float* __restrict__ emb, const int* __restrict__ lab,
    const _Float16* __restrict__ protos, const uint32_t* __restrict__ maskW,
    float* __restrict__ accum, int N, int blocksPerBatch) {
    __shared__ float sAcc[2];
    int b      = blockIdx.x / blocksPerBatch;
    int blkInB = blockIdx.x % blocksPerBatch;
    int t = threadIdx.x, lane = t & 31, wv = t >> 5;
    if (t == 0) { sAcc[0] = 0.0f; sAcc[1] = 0.0f; }
    __syncthreads();

    int ln = lane & 15;
    int hi = (lane >> 4) & 1;   // 0: lanes 0-15 (rows r), 1: lanes 16-31 (rows r+8)

    // B fragments: lane = column (category), 16 contiguous K-halves per lane
    const _Float16* pb = protos + (size_t)b * PADC * DIM;
    v16h bf[5];
    #pragma unroll
    for (int g = 0; g < 5; ++g)
        bf[g] = *(const v16h*)(pb + (g * 16 + ln) * DIM + hi * 16);

    // present bits: column g*16+ln maps to a compile-time word per g
    const uint32_t* mwp = maskW + b * 3;
    uint32_t m0 = mwp[0], m1 = mwp[1], m2 = mwp[2];
    uint32_t wsel[5] = { m0, m0, m1, m1, m2 };
    bool pr[5];
    #pragma unroll
    for (int g = 0; g < 5; ++g)
        pr[g] = (wsel[g] >> (((g & 1) << 4) + ln)) & 1u;

    const float* eb = emb + (size_t)b * N * DIM;
    const int*   lbp = lab + (size_t)b * N;
    int baseA  = hi ? 16 : 0;
    int baseA2 = hi ? 24 : 8;
    int tileBase = (blkInB * 8 + wv) * TPW;

    float lossAcc = 0.0f, cntAcc = 0.0f;
    #pragma unroll 1
    for (int it = 0; it < TPW; ++it) {
        int rowBase = (tileBase + it) * 16;

        // A fragment: lane holds row M = ln; contiguous K = 16*hi + s (matches B)
        const float* er = eb + (size_t)(rowBase + ln) * DIM;
        v4f e0 = *(const v4f*)(er + baseA);
        v4f e1 = *(const v4f*)(er + baseA + 4);
        v4f e2 = *(const v4f*)(er + baseA2);
        v4f e3 = *(const v4f*)(er + baseA2 + 4);
        float sq = 0.0f;
        #pragma unroll
        for (int j = 0; j < 4; ++j)
            sq += e0[j]*e0[j] + e1[j]*e1[j] + e2[j]*e2[j] + e3[j]*e3[j];
        sq += __shfl_xor(sq, 16, 32);    // combine the two K-halves of the row
        // fold 1/T into the normalization: WMMA output is the final logits
        float scale = INV_T / fmaxf(sqrtf(sq), 1e-12f);
        v16h a;
        #pragma unroll
        for (int j = 0; j < 4; ++j) {
            a[j]      = (_Float16)(e0[j] * scale);
            a[4 + j]  = (_Float16)(e1[j] * scale);
            a[8 + j]  = (_Float16)(e2[j] * scale);
            a[12 + j] = (_Float16)(e3[j] * scale);
        }

        v8f cz = {};
        v8f acc[5];
        #pragma unroll
        for (int g = 0; g < 5; ++g)
            acc[g] = __builtin_amdgcn_wmma_f32_16x16x32_f16(
                false, a, false, bf[g], (short)0, cz, false, false);

        #pragma unroll
        for (int r = 0; r < 8; ++r) {
            float v[5];
            #pragma unroll
            for (int g = 0; g < 5; ++g)
                v[g] = pr[g] ? acc[g][r] : NEGINF;
            float m = v[0];
            #pragma unroll
            for (int g = 1; g < 5; ++g) m = fmaxf(m, v[g]);
            #pragma unroll
            for (int d = 1; d < 16; d <<= 1) m = fmaxf(m, __shfl_xor(m, d, 32));
            float s = 0.0f;
            #pragma unroll
            for (int g = 0; g < 5; ++g) s += __expf(v[g] - m);
            #pragma unroll
            for (int d = 1; d < 16; d <<= 1) s += __shfl_xor(s, d, 32);
            float lse = m + __logf(s);

            int row = rowBase + r + hi * 8;
            int L = lbp[row];
            int lc = min(max(L, 0), NUM_C - 1);
            float pick = 0.0f;
            #pragma unroll
            for (int g = 0; g < 5; ++g)
                if (g * 16 + ln == lc) pick = v[g];
            #pragma unroll
            for (int d = 1; d < 16; d <<= 1) pick += __shfl_xor(pick, d, 32);

            bool keep = (L >= 0) && !is_missing(lc);
            if (ln == 0 && keep) { lossAcc += (lse - pick); cntAcc += 1.0f; }
        }
    }
    if (ln == 0) { atomicAdd(&sAcc[0], lossAcc); atomicAdd(&sAcc[1], cntAcc); }
    __syncthreads();
    if (t == 0) {
        atomicAdd(&accum[b * 2 + 0], sAcc[0]);
        atomicAdd(&accum[b * 2 + 1], sAcc[1]);
    }
}

// ---------------- phase 4: combine ----------------
__global__ void k_final(const float* __restrict__ accum, float* __restrict__ out, int B) {
    if (threadIdx.x == 0 && blockIdx.x == 0) {
        float s = 0.0f;
        for (int b = 0; b < B; ++b)
            s += accum[2 * b] / fmaxf(accum[2 * b + 1], 1.0f);
        out[0] = 10.0f * s / (float)B;
    }
}

extern "C" void kernel_launch(void* const* d_in, const int* in_sizes, int n_in,
                              void* d_out, int out_size, void* d_ws, size_t ws_size,
                              hipStream_t stream) {
    const float* emb = (const float*)d_in[0];
    const int*   lab = (const int*)d_in[1];
    float* out = (float*)d_out;

    const int B = 4;
    const int N = in_sizes[1] / B;   // 262144

    // workspace layout
    char* ws = (char*)d_ws;
    size_t off = 0;
    float*    gSums = (float*)(ws + off); off += (size_t)B * NUM_C * DIM * sizeof(float);
    float*    gCnt  = (float*)(ws + off); off += (size_t)B * NUM_C * sizeof(float);
    float*    accum = (float*)(ws + off); off += (size_t)B * 2 * sizeof(float);
    uint32_t* maskW = (uint32_t*)(ws + off); off += (size_t)B * 3 * sizeof(uint32_t);
    size_t zeroWords = off / 4;      // sums + counts + accum + masks, all zero-init
    off = (off + 63) & ~(size_t)63;
    _Float16* protos = (_Float16*)(ws + off); off += (size_t)B * PADC * DIM * sizeof(_Float16);

    k_zero<<<(int)((zeroWords + 255) / 256), 256, 0, stream>>>((float*)ws, (int)zeroWords);

    const int chunks = 256;
    k_segsum<<<B * chunks, 256, 0, stream>>>(emb, lab, gSums, gCnt, N, chunks);

    k_protos<<<B, 256, 0, stream>>>(gSums, gCnt, protos, maskW);

    int tilesPerBatch  = N / 16;                   // 16384
    int blocksPerBatch = tilesPerBatch / (8 * TPW); // 8 waves/block, TPW tiles/wave
    k_loss<<<B * blocksPerBatch, 256, 0, stream>>>(emb, lab, protos, maskW,
                                                   accum, N, blocksPerBatch);

    k_final<<<1, 1, 0, stream>>>(accum, out, B);
}
